// SelfAttention_20675972563111
// MI455X (gfx1250) — matrix operand order repaired and checked by reading, not compiled, not run on hardware
//
#include <hip/hip_runtime.h>
#include <hip/hip_bf16.h>
#include <cstddef>
#include <cstdint>

// ---------------- problem constants (from reference) ----------------
constexpr int BB = 2;      // batch
constexpr int SS = 2048;   // sequence
constexpr int CC = 1024;   // model dim
constexpr int HH = 1024;   // hidden (NH*DH)
constexpr int NH = 16;     // heads
constexpr int DH = 64;     // head dim
constexpr int HALFW = 128; // ATTN_WIDTH/2

// ---------------- WMMA types & helpers ----------------
typedef __attribute__((ext_vector_type(16))) __bf16 v16bf;
typedef __attribute__((ext_vector_type(8)))  float  v8f;
typedef __attribute__((ext_vector_type(4)))  unsigned int v4u;
typedef __attribute__((ext_vector_type(8)))  int v8i_t;
typedef __attribute__((ext_vector_type(4)))  int v4i_t;

#ifndef __has_builtin
#define __has_builtin(x) 0
#endif
#if __has_builtin(__builtin_amdgcn_tensor_load_to_lds) && \
    __has_builtin(__builtin_amdgcn_s_wait_tensorcnt)
#define USE_TDM 1
#else
#define USE_TDM 0
#endif

__device__ __forceinline__ v8f wmma_bf16(v16bf a, v16bf b, v8f c) {
  return __builtin_amdgcn_wmma_f32_16x16x32_bf16(false, a, false, b, (short)0, c,
                                                 false, false);
}

// fp32 -> bf16 hi + bf16 lo (error-compensated "bf16x2" operands)
__device__ __forceinline__ void split_bf16(float f, __bf16& hi, __bf16& lo) {
  unsigned int u = __builtin_bit_cast(unsigned int, f);
  unsigned short hs = (unsigned short)(u >> 16);
  hi = __builtin_bit_cast(__bf16, hs);
  float hf = __builtin_bit_cast(float, (unsigned int)hs << 16);
  unsigned int ru = __builtin_bit_cast(unsigned int, f - hf);
  lo = __builtin_bit_cast(__bf16, (unsigned short)(ru >> 16));
}
__device__ __forceinline__ float bf2f(__bf16 b) {
  unsigned short s = __builtin_bit_cast(unsigned short, b);
  return __builtin_bit_cast(float, (unsigned int)s << 16);
}

// ---------------- Tensor Data Mover: 2D bf16 tile -> LDS ----------------
// D# per CDNA5 ISA: group0 {count=1, lds_addr, global_addr,
// type=2}; group1 {data_size=1(2B), tensor_dim0/1, tile_dim0/1, dim0 stride}.
#if USE_TDM
__device__ __forceinline__ void tdm_load_bf16_2d(const __bf16* gsrc,
                                                 unsigned int lds_byte_off,
                                                 int rows, int cols,
                                                 int row_stride_elems) {
  unsigned long long ga = (unsigned long long)(uintptr_t)gsrc;
  v4u g0;
  g0[0] = 1u;                                   // count=1 (valid descriptor)
  g0[1] = lds_byte_off;                         // lds_addr [63:32]
  g0[2] = (unsigned int)(ga & 0xFFFFFFFFu);     // global_addr lo
  g0[3] = (unsigned int)((ga >> 32) & 0x1FFFFFFu) | (2u << 30); // hi + type=2
  v8i_t g1;
  g1[0] = (int)(1u << 16);                      // workgroup_mask=0, data_size=1
  g1[1] = (int)((unsigned)(cols & 0xFFFF) << 16);          // tensor_dim0 lo
  g1[2] = (int)(((unsigned)cols >> 16) & 0xFFFFu) |
          (int)((unsigned)(rows & 0xFFFF) << 16);          // dim0 hi, dim1 lo
  g1[3] = (int)(((unsigned)rows >> 16) & 0xFFFFu) |
          (int)((unsigned)cols << 16);                     // dim1 hi, tile_dim0
  g1[4] = rows;                                 // tile_dim1 (tile_dim2=0)
  g1[5] = row_stride_elems;                     // tensor_dim0_stride lo32
  g1[6] = 0;
  g1[7] = 0;
  v4i_t z4 = {0, 0, 0, 0};
#if __clang_major__ >= 23
  v8i_t z8 = {0, 0, 0, 0, 0, 0, 0, 0};
  __builtin_amdgcn_tensor_load_to_lds(g0, g1, z4, z4, z8, 0);
#else
  __builtin_amdgcn_tensor_load_to_lds(g0, g1, z4, z4, 0);
#endif
}
#endif

// =====================================================================
// fp32 -> bf16x2 conversions (run once; removes splits from hot loops)
// =====================================================================
__global__ void split_rm_kernel(const float* __restrict__ in,
                                __bf16* __restrict__ hi, __bf16* __restrict__ lo,
                                int n) {
  int i = blockIdx.x * 256 + threadIdx.x;
  if (i < n) { __bf16 h, l; split_bf16(in[i], h, l); hi[i] = h; lo[i] = l; }
}
// W[R][C] -> WT[C][R] bf16 hi/lo (so GEMM B-fragments are contiguous)
__global__ void split_tr_kernel(const float* __restrict__ in,
                                __bf16* __restrict__ hiT, __bf16* __restrict__ loT,
                                int R, int C) {
  int i = blockIdx.x * 256 + threadIdx.x;
  if (i < R * C) {
    int r = i / C, c = i % C;
    __bf16 h, l; split_bf16(in[i], h, l);
    hiT[(size_t)c * R + r] = h; loT[(size_t)c * R + r] = l;
  }
}

// =====================================================================
// GEMM  C[M,N] = A[M,K] * B[K,N], A given row-major bf16 hi/lo, B given
// TRANSPOSED (BT[N][K]) bf16 hi/lo. bf16x2 WMMA (3 wmma per tile-chunk).
// Block: 128 thr (4 waves), tile 64x64, K chunked by 32.
// LDS holds fragment-ready layouts -> whole-v16bf ds loads, no packing VALU.
// Optional epilogues: fp32 C, row-major bf16x2 C, head-transposed bf16x2 C.
// =====================================================================
__global__ __launch_bounds__(128) void gemm_bf16x2(
    const __bf16* __restrict__ Ahi, const __bf16* __restrict__ Alo,
    const __bf16* __restrict__ BThi, const __bf16* __restrict__ BTlo,
    int M, int N, int K,
    float* __restrict__ Cf,
    __bf16* __restrict__ Chi, __bf16* __restrict__ Clo,
    __bf16* __restrict__ CThi, __bf16* __restrict__ CTlo) {
  __shared__ __align__(32) __bf16 Af[2][2048]; // [hi/lo] A-fragment order
  __shared__ __align__(32) __bf16 Bf[2][2048]; // [hi/lo] raw [n][k] == frag order
  const int tid = threadIdx.x;
  const int wave = tid >> 5, lane = tid & 31;
  const int hl = lane >> 4, ml = lane & 15;
  const int m0 = blockIdx.x * 64, n0 = blockIdx.y * 64;

  v8f acc[4] = {};

  for (int k0 = 0; k0 < K; k0 += 32) {
    // ---- stage A chunk: 16B segments permuted into A-fragment order ----
    // seg s of row m covers k=[s*8, s*8+8); frag dest: hl=s>>1, part=s&1.
    for (int u = tid; u < 512; u += 128) {
      int arr = u >> 8, m = (u >> 2) & 63, s = u & 3;
      const __bf16* src = (arr ? Alo : Ahi) + (size_t)(m0 + m) * K + k0 + s * 8;
      __bf16* dst = &Af[arr][((m * 2 + (s >> 1)) * 16) + (s & 1) * 8];
      *(uint4*)dst = *(const uint4*)src;
    }
    // ---- stage BT chunk raw [n][32k] (already fragment order) ----
    for (int u = tid; u < 512; u += 128) {
      int arr = u >> 8, n = (u >> 2) & 63, s = u & 3;
      const __bf16* src = (arr ? BTlo : BThi) + (size_t)(n0 + n) * K + k0 + s * 8;
      *(uint4*)&Bf[arr][n * 32 + s * 8] = *(const uint4*)src;
    }
    if (k0 + 32 < K) {
      __builtin_prefetch(&Ahi[(size_t)(m0 + (tid >> 1)) * K + k0 + 32], 0, 1);
      __builtin_prefetch(&BThi[(size_t)(n0 + (tid >> 1)) * K + k0 + 32], 0, 1);
    }
    __syncthreads();

    v16bf a_hi = *(const v16bf*)&Af[0][((wave * 16 + ml) * 2 + hl) * 16];
    v16bf a_lo = *(const v16bf*)&Af[1][((wave * 16 + ml) * 2 + hl) * 16];
#pragma unroll
    for (int sub = 0; sub < 4; ++sub) {
      v16bf b_hi = *(const v16bf*)&Bf[0][(sub * 16 + ml) * 32 + hl * 16];
      v16bf b_lo = *(const v16bf*)&Bf[1][(sub * 16 + ml) * 32 + hl * 16];
      acc[sub] = wmma_bf16(a_hi, b_hi, acc[sub]);
      acc[sub] = wmma_bf16(a_hi, b_lo, acc[sub]);
      acc[sub] = wmma_bf16(a_lo, b_hi, acc[sub]);
    }
    __syncthreads();
  }

  // D layout: element r -> row = r + 8*hl, col = ml (per 16x16 sub-tile)
#pragma unroll
  for (int sub = 0; sub < 4; ++sub)
#pragma unroll
    for (int r = 0; r < 8; ++r) {
      int row = m0 + wave * 16 + hl * 8 + r;
      int col = n0 + sub * 16 + ml;
      float v = acc[sub][r];
      if (Cf) Cf[(size_t)row * N + col] = v;
      if (Chi) {
        __bf16 h, l; split_bf16(v, h, l);
        Chi[(size_t)row * N + col] = h; Clo[(size_t)row * N + col] = l;
      }
      if (CThi) { // head-transposed V: VT[(b*HH + d)][s]
        int b = row >> 11, s = row & (SS - 1);
        size_t t = ((size_t)b * HH + col) * SS + s;
        __bf16 h, l; split_bf16(v, h, l);
        CThi[t] = h; CTlo[t] = l;
      }
    }
}

// =====================================================================
// Vsum[bh][d] = sum_s V[b,s,h*64+d]  (from head-transposed bf16x2 V)
// =====================================================================
__global__ void vsum_kernel(const __bf16* __restrict__ VThi,
                            const __bf16* __restrict__ VTlo,
                            float* __restrict__ Vsum) {
  int bh = blockIdx.x, b = bh >> 4, h = bh & 15;
  int d = threadIdx.x; // 64
  size_t row = ((size_t)b * HH + h * DH + d) * SS;
  float s = 0.f;
  for (int k = 0; k < SS; ++k) s += bf2f(VThi[row + k]) + bf2f(VTlo[row + k]);
  Vsum[bh * DH + d] = s;
}

// =====================================================================
// Fused banded attention; block = q-tile(16) x (b,h); 4 waves.
//  out_q = (sum_band (exp(e)-1) V + Vsum) / (sum_band (exp(e)-1) + S),
//  e = exp(score/8 - rowmax_full). K-tiles streamed via TDM double buffer.
// =====================================================================
__global__ __launch_bounds__(128) void attn_kernel(
    const __bf16* __restrict__ Qhi, const __bf16* __restrict__ Qlo,
    const __bf16* __restrict__ Khi, const __bf16* __restrict__ Klo,
    const __bf16* __restrict__ VThi, const __bf16* __restrict__ VTlo,
    const float* __restrict__ Vsum,
    __bf16* __restrict__ AOhi, __bf16* __restrict__ AOlo) {
  const int qt = blockIdx.x;  // 0..127
  const int bh = blockIdx.y;  // 0..31
  const int b = bh >> 4, h = bh & 15;
  const int tid = threadIdx.x, wave = tid >> 5, lane = tid & 31;
  const int hl = lane >> 4, ml = lane & 15;

  __shared__ __align__(32) __bf16 qf[2][2][512];       // [hi/lo][chunk] frag order
  __shared__ __align__(32) __bf16 kst[4][2][2][1024];  // [wave][buf][hi/lo] 16x64
  __shared__ float sc[17][16][16];
  __shared__ float mpart[4][16];
  __shared__ float mrow[16];
  __shared__ float denom[16];

  const size_t base = (size_t)b * SS * HH + h * DH;

  // stage Q tile into A-fragment order (16B segs): seg s of row m covers
  // cols [s*8, s*8+8) -> chunk c=s>>2, hl=(s>>1)&1, part=s&1
  for (int u = tid; u < 256; u += 128) {
    int arr = u >> 7, rem = u & 127, m = rem >> 3, s = rem & 7;
    int c = s >> 2, sl = s & 3;
    const __bf16* src = (arr ? Qlo : Qhi) + base + (size_t)(qt * 16 + m) * HH + s * 8;
    *(uint4*)&qf[arr][c][(m * 2 + (sl >> 1)) * 16 + (sl & 1) * 8] =
        *(const uint4*)src;
  }

#if USE_TDM
  { // prime double buffer: this wave's first K tile (hi+lo)
    int kt0 = wave;
    const __bf16* sh = Khi + (size_t)(b * SS + kt0 * 16) * HH + h * DH;
    const __bf16* sl = Klo + (size_t)(b * SS + kt0 * 16) * HH + h * DH;
    tdm_load_bf16_2d(sh, (unsigned int)(uintptr_t)&kst[wave][0][0][0], 16, 64, HH);
    tdm_load_bf16_2d(sl, (unsigned int)(uintptr_t)&kst[wave][0][1][0], 16, 64, HH);
  }
#endif
  __syncthreads();

  v16bf qa_hi[2], qa_lo[2];
#pragma unroll
  for (int c = 0; c < 2; ++c) {
    qa_hi[c] = *(const v16bf*)&qf[0][c][(ml * 2 + hl) * 16];
    qa_lo[c] = *(const v16bf*)&qf[1][c][(ml * 2 + hl) * 16];
  }

  const int w0t = qt - 8;
  float rmax[8];
#pragma unroll
  for (int r = 0; r < 8; ++r) rmax[r] = -3.0e38f;

  // ---- pass 1: full QK^T scan for row max; keep in-window score tiles ----
  int cur = 0;
  for (int kt = wave; kt < SS / 16; kt += 4) {
#if USE_TDM
    if (kt + 4 < SS / 16) { // prefetch next tile into other buffer (async DMA)
      const __bf16* sh = Khi + (size_t)(b * SS + (kt + 4) * 16) * HH + h * DH;
      const __bf16* sl = Klo + (size_t)(b * SS + (kt + 4) * 16) * HH + h * DH;
      tdm_load_bf16_2d(sh, (unsigned int)(uintptr_t)&kst[wave][cur ^ 1][0][0], 16, 64, HH);
      tdm_load_bf16_2d(sl, (unsigned int)(uintptr_t)&kst[wave][cur ^ 1][1][0], 16, 64, HH);
      __builtin_amdgcn_s_wait_tensorcnt(2); // current tile's 2 loads complete
    } else {
      __builtin_amdgcn_s_wait_tensorcnt(0);
    }
#endif
    v8f acc = {};
#pragma unroll
    for (int c = 0; c < 2; ++c) {
#if USE_TDM
      v16bf kb_hi = *(const v16bf*)&kst[wave][cur][0][ml * 64 + c * 32 + hl * 16];
      v16bf kb_lo = *(const v16bf*)&kst[wave][cur][1][ml * 64 + c * 32 + hl * 16];
#else
      size_t koff = base + (size_t)(kt * 16 + ml) * HH + c * 32 + hl * 16;
      v16bf kb_hi = *(const v16bf*)&Khi[koff];
      v16bf kb_lo = *(const v16bf*)&Klo[koff];
#endif
      acc = wmma_bf16(qa_hi[c], kb_hi, acc);
      acc = wmma_bf16(qa_hi[c], kb_lo, acc);
      acc = wmma_bf16(qa_lo[c], kb_hi, acc);
    }
    cur ^= 1;
    int slot = kt - w0t;
    bool inwin = (slot >= 0 && slot <= 16);
#pragma unroll
    for (int r = 0; r < 8; ++r) {
      float s = acc[r] * 0.125f; // 1/sqrt(64)
      rmax[r] = fmaxf(rmax[r], s);
      if (inwin) sc[slot][hl * 8 + r][ml] = s;
    }
  }
  // row-max across the 16 lanes of each half-group, then across waves
#pragma unroll
  for (int r = 0; r < 8; ++r) {
    float v = rmax[r];
    v = fmaxf(v, __shfl_xor(v, 1, 32));
    v = fmaxf(v, __shfl_xor(v, 2, 32));
    v = fmaxf(v, __shfl_xor(v, 4, 32));
    v = fmaxf(v, __shfl_xor(v, 8, 32));
    rmax[r] = v;
  }
  if (ml == 0) {
#pragma unroll
    for (int r = 0; r < 8; ++r) mpart[wave][hl * 8 + r] = rmax[r];
  }
  __syncthreads();
  if (tid < 16)
    mrow[tid] = fmaxf(fmaxf(mpart[0][tid], mpart[1][tid]),
                      fmaxf(mpart[2][tid], mpart[3][tid]));
  __syncthreads();

  // ---- weights: g = exp(exp(s - m)) - 1 on band, else 0 ----
  for (int idx = tid; idx < 17 * 256; idx += 128) {
    int slot = idx >> 8, rem = idx & 255, i = rem >> 4, j = rem & 15;
    int kt = w0t + slot;
    int dd = (kt * 16 + j) - (qt * 16 + i);
    float g = 0.0f;
    if (kt >= 0 && kt < SS / 16 && dd >= -HALFW && dd <= HALFW - 1) {
      float e1 = expf(sc[slot][i][j] - mrow[i]);
      g = expf(e1) - 1.0f;
    }
    sc[slot][i][j] = g;
  }
  __syncthreads();
  if (tid < 16) {
    float s = (float)SS; // out-of-band keys contribute exp(0)=1 each
    for (int kk = 0; kk < 272; ++kk) s += sc[kk >> 4][tid][kk & 15];
    denom[tid] = s;
  }
  __syncthreads();

  // ---- AV over 272-wide window; wave owns dh-tile wave*16 ----
  {
    v8f acc = {};
    const int n0 = wave * 16;
    const size_t vrow = ((size_t)b * HH + h * DH + n0 + ml) * SS;
    for (int c = 0; c < 9; ++c) {
      v16bf pa_hi, pa_lo, vb_hi, vb_lo;
#pragma unroll
      for (int e = 0; e < 16; ++e) {
        int kk = c * 32 + hl * 16 + e; // A element->K map
        float g = (kk < 272) ? sc[kk >> 4][ml][kk & 15] : 0.0f;
        __bf16 hv, lv; split_bf16(g, hv, lv);
        pa_hi[e] = hv; pa_lo[e] = lv;
      }
      int k0c = w0t * 16 + c * 32 + hl * 16;
      if (k0c >= 0 && k0c + 15 < SS) { // fast contiguous path (VT row-major in s)
        vb_hi = *(const v16bf*)&VThi[vrow + k0c];
        vb_lo = *(const v16bf*)&VTlo[vrow + k0c];
      } else {
#pragma unroll
        for (int e = 0; e < 16; ++e) {
          int key = k0c + e;
          key = key < 0 ? 0 : (key > SS - 1 ? SS - 1 : key); // A=0 kills OOB
          vb_hi[e] = VThi[vrow + key];
          vb_lo[e] = VTlo[vrow + key];
        }
      }
      acc = wmma_bf16(pa_hi, vb_hi, acc);
      acc = wmma_bf16(pa_hi, vb_lo, acc);
      acc = wmma_bf16(pa_lo, vb_hi, acc);
    }
#pragma unroll
    for (int r = 0; r < 8; ++r) {
      int i = hl * 8 + r;
      int nn = n0 + ml;
      float v = (acc[r] + Vsum[bh * DH + nn]) / denom[i];
      __bf16 hv, lv; split_bf16(v, hv, lv);
      size_t o = base + (size_t)(qt * 16 + i) * HH + nn;
      AOhi[o] = hv; AOlo[o] = lv;
    }
  }
}

// =====================================================================
// launcher
// =====================================================================
extern "C" void kernel_launch(void* const* d_in, const int* in_sizes, int n_in,
                              void* d_out, int out_size, void* d_ws, size_t ws_size,
                              hipStream_t stream) {
  const float* x  = (const float*)d_in[0];
  const float* Wq = (const float*)d_in[1];
  const float* Wk = (const float*)d_in[2];
  const float* Wv = (const float*)d_in[3];
  const float* Wo = (const float*)d_in[4];
  float* out = (float*)d_out;

  char* p = (char*)d_ws;
  auto abf = [&](size_t elems) { __bf16* r = (__bf16*)p; p += elems * 2; return r; };
  const size_t nBSH = (size_t)BB * SS * HH; // 4M
  const size_t nW = (size_t)CC * HH;        // 1M

  __bf16 *xhi = abf(nBSH), *xlo = abf(nBSH);
  __bf16 *WqThi = abf(nW), *WqTlo = abf(nW);
  __bf16 *WkThi = abf(nW), *WkTlo = abf(nW);
  __bf16 *WvThi = abf(nW), *WvTlo = abf(nW);
  __bf16 *WoThi = abf(nW), *WoTlo = abf(nW);
  __bf16 *QhiB = abf(nBSH), *QloB = abf(nBSH);
  __bf16 *KhiB = abf(nBSH), *KloB = abf(nBSH);
  __bf16 *VThi = abf(nBSH), *VTlo = abf(nBSH);
  __bf16 *AOhi = abf(nBSH), *AOlo = abf(nBSH);
  float* Vs = (float*)p; // + BB*NH*DH floats

  const int M = BB * SS; // 4096
  dim3 blk(128);

  // one-time operand conversions
  split_rm_kernel<<<(int)(nBSH / 256), 256, 0, stream>>>(x, xhi, xlo, (int)nBSH);
  split_tr_kernel<<<(int)(nW / 256), 256, 0, stream>>>(Wq, WqThi, WqTlo, CC, HH);
  split_tr_kernel<<<(int)(nW / 256), 256, 0, stream>>>(Wk, WkThi, WkTlo, CC, HH);
  split_tr_kernel<<<(int)(nW / 256), 256, 0, stream>>>(Wv, WvThi, WvTlo, CC, HH);
  split_tr_kernel<<<(int)(nW / 256), 256, 0, stream>>>(Wo, WoThi, WoTlo, HH, CC);

  dim3 gproj(M / 64, HH / 64);
  // Q, K: bf16x2 row-major outputs. V: head-transposed bf16x2 output.
  gemm_bf16x2<<<gproj, blk, 0, stream>>>(xhi, xlo, WqThi, WqTlo, M, HH, CC,
                                         nullptr, QhiB, QloB, nullptr, nullptr);
  gemm_bf16x2<<<gproj, blk, 0, stream>>>(xhi, xlo, WkThi, WkTlo, M, HH, CC,
                                         nullptr, KhiB, KloB, nullptr, nullptr);
  gemm_bf16x2<<<gproj, blk, 0, stream>>>(xhi, xlo, WvThi, WvTlo, M, HH, CC,
                                         nullptr, nullptr, nullptr, VThi, VTlo);

  vsum_kernel<<<BB * NH, 64, 0, stream>>>(VThi, VTlo, Vs);

  attn_kernel<<<dim3(SS / 16, BB * NH), blk, 0, stream>>>(
      QhiB, QloB, KhiB, KloB, VThi, VTlo, Vs, AOhi, AOlo);

  gemm_bf16x2<<<dim3(M / 64, CC / 64), blk, 0, stream>>>(
      AOhi, AOlo, WoThi, WoTlo, M, CC, HH, out, nullptr, nullptr, nullptr, nullptr);
}
